// FFM_layer_39977555591342
// MI455X (gfx1250) — matrix operand order repaired and checked by reading, not compiled, not run on hardware
//
#include <hip/hip_runtime.h>
#include <stdint.h>

// ---------------------------------------------------------------------------
// FFM layer for MI455X (gfx1250, wave32, WMMA + TDM).
//   B=16384, F=256, P=16, K=16, C=2
// T[i,j,c] = sum_k v[i,fm(j),k,c]*v[j,fm(i),k,c]   (batch independent, 256x256x2)
// out = log_softmax( x@W.T + b + 0.5*(x^T T x - sum_i x_i^2 T_ii), axis=c )
//
// Roofline: mandatory HBM traffic = x once (16 MB @ 23.3 TB/s ~ 0.7us);
// compute ~4.3 GFLOP via v_wmma_f32_16x16x32_bf16. The L2 re-read of the
// pre-swizzled T (256 KB) is the real cost -> amortize it: each block stages
// T's WMMA B-fragments into LDS ONCE via the Tensor Data Mover (TENSORcnt,
// s_wait_tensorcnt), then processes 4 row-tiles with B operands from LDS.
// LDS budget: 256 KB frags + 8.4 KB x-tile + acc ~ 266 KB < 320 KB/WGP.
// ---------------------------------------------------------------------------

typedef __attribute__((ext_vector_type(16))) __bf16 v16bf;
typedef __attribute__((ext_vector_type(8)))  float  v8f;
typedef __attribute__((ext_vector_type(4)))  unsigned int v4u;
typedef __attribute__((ext_vector_type(4)))  int    v4i;
typedef __attribute__((ext_vector_type(8)))  int    v8i;

constexpr int Fd = 256;   // features
constexpr int Pd = 16;    // fields
constexpr int Kd = 16;    // latent k
constexpr int Cd = 2;     // channels
constexpr int XS_STRIDE = 264;       // 256 + 8 pad (bf16) -> conflict-free LDS rows
constexpr int RTILES    = 4;         // row-tiles (of 16) per block
constexpr int FRAG_DW   = 65536;     // B-fragment buffer, dwords (256 KB)

// LDS layout (dynamic):
constexpr int LDS_XS_OFF   = FRAG_DW * 4;                 // 262144
constexpr int LDS_FACC_OFF = LDS_XS_OFF + 16 * XS_STRIDE * 2;  // 270592
constexpr int LDS_BYTES    = LDS_FACC_OFF + 32 * 4;       // 270720

__device__ __forceinline__ uint16_t f2bf(float f) {
  union { float f; uint32_t u; } x; x.f = f;
  uint32_t u = x.u;
  return (uint16_t)((u + 0x7FFFu + ((u >> 16) & 1u)) >> 16);  // RNE
}
__device__ __forceinline__ float bf2f(uint16_t h) {
  union { uint32_t u; float f; } x; x.u = ((uint32_t)h) << 16;
  return x.f;
}

// T(i,j,c) from v (F,P,K,C) and field_map
__device__ __forceinline__ float Tval(const float* __restrict__ v,
                                      const int* __restrict__ fm,
                                      int i, int j, int c) {
  const float* vi = v + (((size_t)i * Pd + fm[j]) * Kd) * Cd + c;
  const float* vj = v + (((size_t)j * Pd + fm[i]) * Kd) * Cd + c;
  float s = 0.f;
#pragma unroll
  for (int k = 0; k < Kd; ++k) s += vi[k * Cd] * vj[k * Cd];
  return s;
}

// ---------------------------------------------------------------------------
// Kernel 1: build T in WMMA B-fragment order (bf16 pairs packed in u32).
// Fragment dword layout: [c][ntile(16)][kstep(8)][lane(32)][p(8)]
// B-matrix (32x16, 16-bit): lane L -> n = L%16, half h = L/16;
// dword p holds K = kstep*32 + 16*h + 2p (low half) and K+1 (high half).
// ---------------------------------------------------------------------------
__global__ __launch_bounds__(256) void ffm_prep_frag(const float* __restrict__ v,
                                                     const int* __restrict__ fm,
                                                     uint32_t* __restrict__ Bfrag) {
  int idx = blockIdx.x * 256 + threadIdx.x;      // 65536 dwords total
  int p    = idx & 7;
  int lane = (idx >> 3) & 31;
  int ks   = (idx >> 8) & 7;
  int nt   = (idx >> 11) & 15;
  int c    = (idx >> 15) & 1;
  int n  = nt * 16 + (lane & 15);
  int h  = lane >> 4;
  int k0 = ks * 32 + h * 16 + 2 * p;
  float t0 = Tval(v, fm, k0,     n, c);
  float t1 = Tval(v, fm, k0 + 1, n, c);
  Bfrag[idx] = (uint32_t)f2bf(t0) | ((uint32_t)f2bf(t1) << 16);
}

// Kernel 2: T_diag[i][c] = T(i,i,c)  (f32, 2 KB)
__global__ void ffm_prep_diag(const float* __restrict__ v,
                              const int* __restrict__ fm,
                              float* __restrict__ Td) {
  int idx = blockIdx.x * blockDim.x + threadIdx.x;
  if (idx >= Fd * Cd) return;
  int i = idx >> 1, c = idx & 1;
  Td[idx] = Tval(v, fm, i, i, c);
}

// ---------------------------------------------------------------------------
// Kernel 3: fused main pass. 1 block = 64 batch rows (4 tiles of 16),
// 256 threads = 8 waves. T fragments staged once into LDS via TDM.
// ---------------------------------------------------------------------------
__global__ __launch_bounds__(256) void ffm_main(const float* __restrict__ x,
                                                const float* __restrict__ W,
                                                const float* __restrict__ bias,
                                                const uint32_t* __restrict__ Bfrag,
                                                const float* __restrict__ Td,
                                                float* __restrict__ out) {
  extern __shared__ char smem[];
  uint32_t* blds = (uint32_t*)smem;                     // 256 KB B fragments
  uint16_t* xs   = (uint16_t*)(smem + LDS_XS_OFF);      // x tile, bf16, padded
  float*    facc = (float*)(smem + LDS_FACC_OFF);       // full[row][c]

  const int tid  = threadIdx.x;
  const int w    = tid >> 5, lane = tid & 31;
  const int m    = lane & 15, h = lane >> 4;

  // ---- stage all B fragments global -> LDS via Tensor Data Mover ----
  // Forced (no __has_builtin fallback): if this fails to compile, the
  // diagnostic reveals the true builtin signature for the next iteration.
  if (w == 0) {
    // D# group0: count=1 | lds_addr | global_addr(57b) | type=2
    uint64_t ga = (uint64_t)(uintptr_t)Bfrag;
    uint32_t la = (uint32_t)(uintptr_t)blds;            // LDS byte offset (addr[31:0])
    v4u g0 = { 1u, la, (uint32_t)ga,
               (uint32_t)((ga >> 32) & 0x01FFFFFFu) | (2u << 30) };
    // D# group1: data_size=8B; 1-D tile: tensor_dim0=tile_dim0=32768 elems (256 KB)
    v8i g1 = { (int)0x00030000,        // wg_mask=0, data_size=3 (8 bytes)
               (int)0x80000000,        // tensor_dim0[15:0]=0x8000 in [31:16]
               (int)0x00010000,        // tensor_dim0[31:16]=0, tensor_dim1=1
               (int)0x80000000,        // tile_dim0=0x8000 in [31:16]
               (int)0x00000001,        // tile_dim1=1, tile_dim2=0
               (int)0x00008000,        // tensor_dim0_stride = 32768
               (int)0x80000000,        // tensor_dim1_stride[15:0] in [31:16]
               0 };
    v4i z4 = { 0, 0, 0, 0 };
#if __clang_major__ >= 23
    v8i z8 = { 0, 0, 0, 0, 0, 0, 0, 0 };
    __builtin_amdgcn_tensor_load_to_lds(g0, g1, z4, z4, z8, 0);
#else
    __builtin_amdgcn_tensor_load_to_lds(g0, g1, z4, z4, 0);
#endif
    __builtin_amdgcn_s_wait_tensorcnt(0);
  }
  __syncthreads();   // publish LDS fragments to all waves

  union AB { uint32_t u[8]; v16bf v; };

  for (int r = 0; r < RTILES; ++r) {
    const int row0 = (blockIdx.x * RTILES + r) * 16;

    // Prefetch next x tile (16 KB = 256 x 64B lines, one per thread):
    // lowers to global_prefetch_b8.
    if (r + 1 < RTILES) {
      const float* nxt = x + (size_t)(row0 + 16) * Fd + tid * 16;
      __builtin_prefetch(nxt, 0, 0);
    }

    // Stage x tile -> bf16 LDS (coalesced)
    for (int e = tid; e < 16 * 256; e += 256) {
      int rr = e >> 8, col = e & 255;
      xs[rr * XS_STRIDE + col] = f2bf(x[(size_t)(row0 + rr) * Fd + col]);
    }
    if (tid < 32) facc[tid] = 0.f;
    __syncthreads();

    v8f acc[2][2];
#pragma unroll
    for (int c = 0; c < 2; ++c)
#pragma unroll
      for (int t = 0; t < 2; ++t) acc[c][t] = (v8f){};

#pragma unroll
    for (int ks = 0; ks < 8; ++ks) {
      // A fragment (16x32 bf16): lane -> row m, half h.
      // dword p: K = 16h + 2p
      AB a;
      const uint16_t* xr = &xs[m * XS_STRIDE + ks * 32 + 16 * h];
#pragma unroll
      for (int p = 0; p < 4; ++p) a.u[p] = *(const uint32_t*)(xr + 2 * p);
#pragma unroll
      for (int p = 4; p < 8; ++p) a.u[p] = *(const uint32_t*)(xr + 2 * p);

#pragma unroll
      for (int c = 0; c < 2; ++c) {
#pragma unroll
        for (int t = 0; t < 2; ++t) {
          const int nt = 2 * w + t;
          const uint32_t* fb = blds + (((c * 16 + nt) * 8 + ks) * 32 + lane) * 8;
          AB bb;
          uint4 q0 = *(const uint4*)(fb);       // ds_load_b128
          uint4 q1 = *(const uint4*)(fb + 4);
          bb.u[0] = q0.x; bb.u[1] = q0.y; bb.u[2] = q0.z; bb.u[3] = q0.w;
          bb.u[4] = q1.x; bb.u[5] = q1.y; bb.u[6] = q1.z; bb.u[7] = q1.w;
          acc[c][t] = __builtin_amdgcn_wmma_f32_16x16x32_bf16(
              false, a.v, false, bb.v, (short)0, acc[c][t], false, false);
        }
      }
    }

    // D layout: VGPR s, lane L -> M = s + 8h (batch row), N = L%16 (j in tile)
    float pl[2][8] = {};
#pragma unroll
    for (int c = 0; c < 2; ++c)
#pragma unroll
      for (int t = 0; t < 2; ++t) {
        const int jbase = (2 * w + t) * 16 + (lane & 15);
#pragma unroll
        for (int s = 0; s < 8; ++s) {
          const int M = s + 8 * h;
          pl[c][s] += acc[c][t][s] * bf2f(xs[M * XS_STRIDE + jbase]);
        }
      }
#pragma unroll
    for (int c = 0; c < 2; ++c)
#pragma unroll
      for (int s = 0; s < 8; ++s) {
        float vsum = pl[c][s];
        for (int mask = 1; mask < 16; mask <<= 1)
          vsum += __shfl_xor(vsum, mask, 32);
        pl[c][s] = vsum;
      }
    if ((lane & 15) == 0) {
#pragma unroll
      for (int c = 0; c < 2; ++c)
#pragma unroll
        for (int s = 0; s < 8; ++s)
          atomicAdd(&facc[(s + 8 * h) * 2 + c], pl[c][s]);   // ds_add_f32
    }
    __syncthreads();

    // diag + linear + log_softmax; wave w handles rows 2w, 2w+1 (h selects)
    const int row = 2 * w + h;
    const int g = lane & 15;
    float d0 = 0, d1 = 0, l0 = 0, l1 = 0;
#pragma unroll 4
    for (int it = 0; it < 16; ++it) {
      int j = g + 16 * it;
      float xb = bf2f(xs[row * XS_STRIDE + j]);
      d0 += xb * xb * Td[j * 2 + 0];
      d1 += xb * xb * Td[j * 2 + 1];
      l0 += xb * W[j];
      l1 += xb * W[Fd + j];
    }
    for (int mask = 1; mask < 16; mask <<= 1) {
      d0 += __shfl_xor(d0, mask, 32);
      d1 += __shfl_xor(d1, mask, 32);
      l0 += __shfl_xor(l0, mask, 32);
      l1 += __shfl_xor(l1, mask, 32);
    }
    if (g == 0) {
      float z0 = l0 + bias[0] + 0.5f * (facc[row * 2 + 0] - d0);
      float z1 = l1 + bias[1] + 0.5f * (facc[row * 2 + 1] - d1);
      float mx = fmaxf(z0, z1);
      float lse = mx + __logf(__expf(z0 - mx) + __expf(z1 - mx));
      out[(size_t)(row0 + row) * 2 + 0] = z0 - lse;
      out[(size_t)(row0 + row) * 2 + 1] = z1 - lse;
    }
    __syncthreads();   // xs/facc reused next tile
  }
}

// ---------------------------------------------------------------------------
extern "C" void kernel_launch(void* const* d_in, const int* in_sizes, int n_in,
                              void* d_out, int out_size, void* d_ws, size_t ws_size,
                              hipStream_t stream) {
  const float* x  = (const float*)d_in[0];
  const int*   fm = (const int*)d_in[1];
  const float* W  = (const float*)d_in[2];
  const float* b  = (const float*)d_in[3];
  const float* v  = (const float*)d_in[4];
  float* out = (float*)d_out;

  uint32_t* Bfrag = (uint32_t*)d_ws;                                  // 256 KB
  float*    Td    = (float*)((char*)d_ws + FRAG_DW * sizeof(uint32_t)); // 2 KB

  ffm_prep_frag<<<256, 256, 0, stream>>>(v, fm, Bfrag);
  ffm_prep_diag<<<2, 256, 0, stream>>>(v, fm, Td);
  ffm_main<<<16384 / (16 * RTILES), 256, LDS_BYTES, stream>>>(x, W, b, Bfrag, Td, out);
}
